// GRU_38448547234793
// MI455X (gfx1250) — matrix operand order repaired and checked by reading, not compiled, not run on hardware
//
#include <hip/hip_runtime.h>

// ---------------------------------------------------------------------------
// CDNA5 (gfx1250) GRU: bf16 WMMA GEMMs + LDS-resident recurrent scan
// ---------------------------------------------------------------------------

typedef __attribute__((ext_vector_type(16))) __bf16 v16bf;
typedef __attribute__((ext_vector_type(8)))  float  v8f;

union FragBF {
  v16bf v;
  uint4 u[2];
};

union F8 {
  float4 q[2];
  float  f[8];
};

__device__ __forceinline__ unsigned short f2bf(float f) {
  unsigned int u = __float_as_uint(f);
  u += 0x7FFFu + ((u >> 16) & 1u);          // round-to-nearest-even
  return (unsigned short)(u >> 16);
}
__device__ __forceinline__ float bf2f(unsigned int s) {
  return __uint_as_float(s << 16);
}
__device__ __forceinline__ float sigm(float x) {
  return 1.0f / (1.0f + __expf(-x));
}
__device__ __forceinline__ float tanh_fast(float x) {
  return 1.0f - 2.0f / (__expf(2.0f * x) + 1.0f);
}

__device__ __forceinline__ F8 ldf8(const float* p) {
  F8 r;
  r.q[0] = *(const float4*)(p);
  r.q[1] = *(const float4*)(p + 4);
  return r;
}
__device__ __forceinline__ void bf16x8_load(const unsigned short* p, float* f) {
  const uint4 u = *(const uint4*)p;
  f[0] = bf2f(u.x & 0xFFFFu); f[1] = bf2f(u.x >> 16);
  f[2] = bf2f(u.y & 0xFFFFu); f[3] = bf2f(u.y >> 16);
  f[4] = bf2f(u.z & 0xFFFFu); f[5] = bf2f(u.z >> 16);
  f[6] = bf2f(u.w & 0xFFFFu); f[7] = bf2f(u.w >> 16);
}
__device__ __forceinline__ uint4 bf16x8_pack(const float* f) {
  uint4 u;
  u.x = (unsigned)f2bf(f[0]) | ((unsigned)f2bf(f[1]) << 16);
  u.y = (unsigned)f2bf(f[2]) | ((unsigned)f2bf(f[3]) << 16);
  u.z = (unsigned)f2bf(f[4]) | ((unsigned)f2bf(f[5]) << 16);
  u.w = (unsigned)f2bf(f[6]) | ((unsigned)f2bf(f[7]) << 16);
  return u;
}

// ---------------------------------------------------------------------------
// f32 -> bf16 conversion (grid-stride)
// ---------------------------------------------------------------------------
__global__ void f32_to_bf16_kernel(const float* __restrict__ src,
                                   unsigned short* __restrict__ dst,
                                   long long n) {
  long long i = (long long)blockIdx.x * blockDim.x + threadIdx.x;
  long long stride = (long long)gridDim.x * blockDim.x;
  for (; i < n; i += stride) dst[i] = f2bf(src[i]);
}

// ---------------------------------------------------------------------------
// x_all = Xbf [M,512] @ Wih^T (Wih row-major [1536,512]) -> bf16 [M,1536]
// Block: 256 threads = 8 waves; wave computes 16x64 (4 N-tiles).
// Grid: (M/128, 1536/64)
// ---------------------------------------------------------------------------
__global__ __launch_bounds__(256) void gemm_xall_kernel(
    const unsigned short* __restrict__ A,   // [M,512] bf16
    const unsigned short* __restrict__ W,   // [1536,512] bf16
    unsigned short* __restrict__ C) {       // [M,1536] bf16
  const int K = 512;
  const int N = 1536;
  const int wave = threadIdx.x >> 5;
  const int lane = threadIdx.x & 31;
  const int row  = lane & 15;
  const int half = lane >> 4;

  const int m0 = (blockIdx.x * 8 + wave) * 16;
  const int n0 = blockIdx.y * 64;

  v8f acc[4] = {};

  const unsigned short* ap = A + (size_t)(m0 + row) * K + half * 8;
  const unsigned short* wp = W + (size_t)(n0 + row) * K + half * 8;

  for (int k0 = 0; k0 < K; k0 += 32) {
    FragBF fa;
    fa.u[0] = *(const uint4*)(ap + k0);
    fa.u[1] = *(const uint4*)(ap + k0 + 16);
#pragma unroll
    for (int j = 0; j < 4; ++j) {
      FragBF fb;
      const unsigned short* q = wp + (size_t)j * 16 * K + k0;
      fb.u[0] = *(const uint4*)(q);
      fb.u[1] = *(const uint4*)(q + 16);
      acc[j] = __builtin_amdgcn_wmma_f32_16x16x32_bf16(
          false, fa.v, false, fb.v, (short)0, acc[j], false, false);
    }
  }

#pragma unroll
  for (int j = 0; j < 4; ++j) {
#pragma unroll
    for (int g = 0; g < 8; ++g) {
      C[(size_t)(m0 + g + 8 * half) * N + n0 + j * 16 + row] = f2bf(acc[j][g]);
    }
  }
}

// ---------------------------------------------------------------------------
// Persistent GRU scan. 4 workgroups, each owns 16 batch rows, 1024 threads
// (32 waves). h lives in LDS (bf16). Per step:
//   G[16,1536] = h @ Whh^T   (96 N-tiles -> 3 per wave, 48 WMMAs/wave,
//                             software-pipelined k loop, B streamed from L2)
//   gates + h update (vectorized elementwise over 16x512),
//   h -> LDS + hs (bf16, global), prefetch next x_all row.
// ---------------------------------------------------------------------------
#define LDH 520    // h row stride (bf16 elems), keeps 16B alignment
#define LDG 1544   // G row stride (f32 elems)

__global__ __launch_bounds__(1024) void gru_scan_kernel(
    const unsigned short* __restrict__ Whh,   // [1536,512] bf16
    const unsigned short* __restrict__ Xall,  // [B*T,1536] bf16, row = b*T+t
    const float* __restrict__ bih,            // [1536]
    const float* __restrict__ bhh,            // [1536]
    unsigned short* __restrict__ Hs) {        // [B*T,512] bf16
  const int T = 2048, H = 512, K = 512;

  extern __shared__ char smem[];
  unsigned short* hbf = (unsigned short*)smem;                          // [16][LDH]
  float* Gs   = (float*)(smem + 16 * LDH * sizeof(unsigned short));     // [16][LDG]
  float* brz  = Gs + 16 * LDG;    // [1024]  bih+bhh for r (0..511), z (512..1023)
  float* bnih = brz + 1024;       // [512]   bih for n
  float* bnhh = bnih + 512;       // [512]   bhh for n

  const int tid  = threadIdx.x;
  const int wave = tid >> 5;
  const int lane = tid & 31;
  const int row  = lane & 15;
  const int half = lane >> 4;
  const int b0   = blockIdx.x * 16;

  // zero h0, stage biases in LDS
  for (int i = tid; i < 16 * LDH; i += 1024) hbf[i] = 0;
  brz[tid] = bih[tid] + bhh[tid];                 // tid in [0,1024): r + z gates
  if (tid < 512) {
    bnih[tid] = bih[1024 + tid];
    bnhh[tid] = bhh[1024 + tid];
  }
  __syncthreads();

  // wave-invariant pointers
  const int nt0 = wave * 3;
  const unsigned short* ap  = hbf + row * LDH + half * 8;
  const unsigned short* wp0 = Whh + (size_t)(nt0 * 16 + row) * K + half * 8;
  const unsigned short* wp1 = wp0 + (size_t)16 * K;
  const unsigned short* wp2 = wp1 + (size_t)16 * K;
  float* gbase = Gs + (8 * half) * LDG + row;

  // elementwise mapping: thread -> (r, 8 cols starting at c0)
  const int er = tid >> 6;
  const int c0 = (tid & 63) * 8;

  for (int t = 0; t < T; ++t) {
    // ---- GEMM: G = h @ Whh^T (prefetch-distance-1 software pipeline) ----
    v8f acc0 = {}, acc1 = {}, acc2 = {};
    FragBF fa, fb0, fb1, fb2;
    fa.u[0]  = *(const uint4*)(ap);
    fa.u[1]  = *(const uint4*)(ap + 16);
    fb0.u[0] = *(const uint4*)(wp0);
    fb0.u[1] = *(const uint4*)(wp0 + 16);
    fb1.u[0] = *(const uint4*)(wp1);
    fb1.u[1] = *(const uint4*)(wp1 + 16);
    fb2.u[0] = *(const uint4*)(wp2);
    fb2.u[1] = *(const uint4*)(wp2 + 16);
#pragma unroll
    for (int k0 = 0; k0 < K; k0 += 32) {
      FragBF fan, fbn0, fbn1, fbn2;
      if (k0 + 32 < K) {
        const int kn = k0 + 32;
        fan.u[0]  = *(const uint4*)(ap + kn);
        fan.u[1]  = *(const uint4*)(ap + kn + 16);
        fbn0.u[0] = *(const uint4*)(wp0 + kn);
        fbn0.u[1] = *(const uint4*)(wp0 + kn + 16);
        fbn1.u[0] = *(const uint4*)(wp1 + kn);
        fbn1.u[1] = *(const uint4*)(wp1 + kn + 16);
        fbn2.u[0] = *(const uint4*)(wp2 + kn);
        fbn2.u[1] = *(const uint4*)(wp2 + kn + 16);
      }
      acc0 = __builtin_amdgcn_wmma_f32_16x16x32_bf16(
          false, fa.v, false, fb0.v, (short)0, acc0, false, false);
      acc1 = __builtin_amdgcn_wmma_f32_16x16x32_bf16(
          false, fa.v, false, fb1.v, (short)0, acc1, false, false);
      acc2 = __builtin_amdgcn_wmma_f32_16x16x32_bf16(
          false, fa.v, false, fb2.v, (short)0, acc2, false, false);
      if (k0 + 32 < K) {
        fa = fan; fb0 = fbn0; fb1 = fbn1; fb2 = fbn2;
      }
    }
#pragma unroll
    for (int g = 0; g < 8; ++g) {
      gbase[(size_t)g * LDG + (nt0 + 0) * 16] = acc0[g];
      gbase[(size_t)g * LDG + (nt0 + 1) * 16] = acc1[g];
      gbase[(size_t)g * LDG + (nt0 + 2) * 16] = acc2[g];
    }
    __syncthreads();   // all WMMAs done (h reads finished), G visible

    // ---- gates + h update (vectorized b128 accesses) ----
    {
      const size_t rowIdx = ((size_t)(b0 + er) * T + t);
      const unsigned short* xrow = Xall + rowIdx * 1536;
      const float* grow = Gs + er * LDG;
      unsigned short* hrow = hbf + er * LDH;
      unsigned short* hsrow = Hs + rowIdx * H;

      float xr[8], xz[8], xn[8], hp[8], ho[8];
      bf16x8_load(xrow + c0,        xr);
      bf16x8_load(xrow + 512 + c0,  xz);
      bf16x8_load(xrow + 1024 + c0, xn);
      bf16x8_load(hrow + c0,        hp);
      const F8 hr = ldf8(grow + c0);
      const F8 hz = ldf8(grow + 512 + c0);
      const F8 hn = ldf8(grow + 1024 + c0);
      const F8 br = ldf8(brz + c0);
      const F8 bz = ldf8(brz + 512 + c0);
      const F8 bi = ldf8(bnih + c0);
      const F8 bh = ldf8(bnhh + c0);

      // prefetch next timestep's x_all row (cold HBM stream, one step of lead)
      if (t + 1 < T) {
        const unsigned short* xnext = Xall + (rowIdx + 1) * 1536;
        __builtin_prefetch(xnext + c0, 0, 0);
        __builtin_prefetch(xnext + 512 + c0, 0, 0);
        __builtin_prefetch(xnext + 1024 + c0, 0, 0);
      }

#pragma unroll
      for (int i = 0; i < 8; ++i) {
        const float rv = sigm(xr[i] + hr.f[i] + br.f[i]);
        const float zv = sigm(xz[i] + hz.f[i] + bz.f[i]);
        const float nv = tanh_fast(xn[i] + bi.f[i] + rv * (hn.f[i] + bh.f[i]));
        ho[i] = nv + zv * (hp[i] - nv);     // (1-z)*n + z*h
      }
      const uint4 pk = bf16x8_pack(ho);
      *(uint4*)(hrow + c0) = pk;
      *(uint4*)(hsrow + c0) = pk;
    }
    __syncthreads();   // new h visible before next step's GEMM
  }
}

// ---------------------------------------------------------------------------
// out = hs [M,512] @ proj_w^T (pw row-major [512,512]) + proj_b  -> f32 [M,512]
// Grid: (M/128, 512/64), block 256.
// ---------------------------------------------------------------------------
__global__ __launch_bounds__(256) void proj_kernel(
    const unsigned short* __restrict__ A,   // [M,512] bf16
    const unsigned short* __restrict__ W,   // [512,512] bf16
    const float* __restrict__ bias,         // [512]
    float* __restrict__ Out) {              // [M,512] f32
  const int K = 512;
  const int N = 512;
  const int wave = threadIdx.x >> 5;
  const int lane = threadIdx.x & 31;
  const int row  = lane & 15;
  const int half = lane >> 4;

  const int m0 = (blockIdx.x * 8 + wave) * 16;
  const int n0 = blockIdx.y * 64;

  v8f acc[4] = {};

  const unsigned short* ap = A + (size_t)(m0 + row) * K + half * 8;
  const unsigned short* wp = W + (size_t)(n0 + row) * K + half * 8;

  for (int k0 = 0; k0 < K; k0 += 32) {
    FragBF fa;
    fa.u[0] = *(const uint4*)(ap + k0);
    fa.u[1] = *(const uint4*)(ap + k0 + 16);
#pragma unroll
    for (int j = 0; j < 4; ++j) {
      FragBF fb;
      const unsigned short* q = wp + (size_t)j * 16 * K + k0;
      fb.u[0] = *(const uint4*)(q);
      fb.u[1] = *(const uint4*)(q + 16);
      acc[j] = __builtin_amdgcn_wmma_f32_16x16x32_bf16(
          false, fa.v, false, fb.v, (short)0, acc[j], false, false);
    }
  }

#pragma unroll
  for (int j = 0; j < 4; ++j) {
    const float bv = bias[n0 + j * 16 + row];
#pragma unroll
    for (int g = 0; g < 8; ++g) {
      Out[(size_t)(m0 + g + 8 * half) * N + n0 + j * 16 + row] = acc[j][g] + bv;
    }
  }
}

// ---------------------------------------------------------------------------
// host launcher
// ---------------------------------------------------------------------------
extern "C" void kernel_launch(void* const* d_in, const int* in_sizes, int n_in,
                              void* d_out, int out_size, void* d_ws, size_t ws_size,
                              hipStream_t stream) {
  (void)in_sizes; (void)n_in; (void)out_size; (void)ws_size;

  const float* x   = (const float*)d_in[0];   // [64,2048,512]
  const float* wih = (const float*)d_in[1];   // [1536,512]
  const float* whh = (const float*)d_in[2];   // [1536,512]
  const float* bih = (const float*)d_in[3];   // [1536]
  const float* bhh = (const float*)d_in[4];   // [1536]
  const float* pw  = (const float*)d_in[5];   // [512,512]
  const float* pb  = (const float*)d_in[6];   // [512]
  float* out = (float*)d_out;                 // [64,2048,512]

  const long long B = 64, T = 2048, I = 512, H = 512, G3 = 1536;
  const long long M = B * T;

  char* ws = (char*)d_ws;
  unsigned short* xbf  = (unsigned short*)ws; ws += (size_t)M * I * 2;    // 128 MB
  unsigned short* wihb = (unsigned short*)ws; ws += (size_t)G3 * I * 2;   // 1.5 MB
  unsigned short* whhb = (unsigned short*)ws; ws += (size_t)G3 * H * 2;   // 1.5 MB
  unsigned short* pwb  = (unsigned short*)ws; ws += (size_t)H * H * 2;    // 0.5 MB
  unsigned short* xall = (unsigned short*)ws; ws += (size_t)M * G3 * 2;   // 384 MB
  unsigned short* hsb  = (unsigned short*)ws; ws += (size_t)M * H * 2;    // 128 MB

  // 1) convert to bf16
  f32_to_bf16_kernel<<<4096, 256, 0, stream>>>(x,   xbf,  M * I);
  f32_to_bf16_kernel<<<1536, 256, 0, stream>>>(wih, wihb, G3 * I);
  f32_to_bf16_kernel<<<1536, 256, 0, stream>>>(whh, whhb, G3 * H);
  f32_to_bf16_kernel<<<512,  256, 0, stream>>>(pw,  pwb,  H * H);

  // 2) x_all = x @ Wih^T   (bias folded into scan elementwise)
  dim3 g1((unsigned)(M / 128), (unsigned)(G3 / 64));
  gemm_xall_kernel<<<g1, 256, 0, stream>>>(xbf, wihb, xall);

  // 3) recurrent scan: 4 persistent workgroups, 16 batch rows each
  const size_t smem = 16 * LDH * sizeof(unsigned short)
                    + 16 * LDG * sizeof(float)
                    + (1024 + 512 + 512) * sizeof(float);
  gru_scan_kernel<<<4, 1024, smem, stream>>>(whhb, xall, bih, bhh, hsb);

  // 4) out = hs @ proj_w^T + proj_b
  dim3 g2((unsigned)(M / 128), (unsigned)(H / 64));
  proj_kernel<<<g2, 256, 0, stream>>>(hsb, pwb, pb, out);
}
